// KeyframeDiscriminator_18365280158172
// MI455X (gfx1250) — matrix operand order repaired and checked by reading, not compiled, not run on hardware
//
#include <hip/hip_runtime.h>

// ---------------------------------------------------------------------------
// CDNA5 (gfx1250): WMMA f16 GEMMs fed by the Tensor Data Mover (TDM).
//  - gemm: TDM double-buffered A/B f32 tiles -> v_wmma_f32_16x16x32_f16
//  - attention: per-wave TDM K-chunk / rel-window tiles (TENSORcnt is
//    per-wave, so the score loop needs no block barriers at all)
// ---------------------------------------------------------------------------

typedef __attribute__((ext_vector_type(16))) _Float16 v16h;
typedef __attribute__((ext_vector_type(8)))  _Float16 v8h;
typedef __attribute__((ext_vector_type(8)))  float    v8f;
typedef __attribute__((ext_vector_type(4)))  unsigned int u32x4;
typedef __attribute__((ext_vector_type(8)))  int      i32x8;
typedef __attribute__((ext_vector_type(4)))  int      i32x4;

#define ACT_NONE 0
#define ACT_SILU 1
#define ACT_RELU 2
#define ACT_SIG  3

#if defined(__has_builtin)
#if __has_builtin(__builtin_amdgcn_tensor_load_to_lds) && __has_builtin(__builtin_amdgcn_s_wait_tensorcnt)
#define HAVE_TDM 1
#endif
#endif
#ifndef HAVE_TDM
#define HAVE_TDM 0
#endif

__device__ __forceinline__ v8f wmma16(v16h a, v16h b, v8f c){
  return __builtin_amdgcn_wmma_f32_16x16x32_f16(false, a, false, b, (short)0, c, false, false);
}

// ---- WMMA fragment builders (this code's contiguous k order, same for A and B) ----
// A (16x32): lane<16 -> row=lane, K {0..15}; lane>=16 -> K {16..31}.
__device__ __forceinline__ v16h ldsA_frag_h(const _Float16* base, int ld, int row_off, int k0){
  int lane = threadIdx.x & 31; int hs = lane >> 4; int r = lane & 15;
  const _Float16* p = base + (size_t)(row_off + r) * ld + k0 + hs * 16;
  union { v16h v; v8h h[2]; } u;
  u.h[0] = *(const v8h*)p;
  u.h[1] = *(const v8h*)(p + 8);
  return u.v;
}
// A fragment from f32 LDS: explicit float4 derefs -> 4x ds_load_b128, then cvt.
__device__ __forceinline__ v16h ldsA_frag_f32(const float* base, int ld, int row_off, int k0){
  int lane = threadIdx.x & 31; int hs = lane >> 4; int r = lane & 15;
  const float* p = base + (size_t)(row_off + r) * ld + k0 + hs * 16;
  float4 f0 = *(const float4*)p;
  float4 f1 = *(const float4*)(p + 4);
  float4 f2 = *(const float4*)(p + 8);
  float4 f3 = *(const float4*)(p + 12);
  v16h a;
  a[0]=(_Float16)f0.x;  a[1]=(_Float16)f0.y;  a[2]=(_Float16)f0.z;  a[3]=(_Float16)f0.w;
  a[4]=(_Float16)f1.x;  a[5]=(_Float16)f1.y;  a[6]=(_Float16)f1.z;  a[7]=(_Float16)f1.w;
  a[8]=(_Float16)f2.x;  a[9]=(_Float16)f2.y;  a[10]=(_Float16)f2.z; a[11]=(_Float16)f2.w;
  a[12]=(_Float16)f3.x; a[13]=(_Float16)f3.y; a[14]=(_Float16)f3.z; a[15]=(_Float16)f3.w;
  return a;
}
// B (32x16) from transposed-[n][k] f16 source: 16 contiguous halves per lane.
__device__ __forceinline__ v16h ldsBT_frag(const _Float16* base, int ld, int col_off, int k0){
  int lane = threadIdx.x & 31; int hs = lane >> 4; int c = lane & 15;
  const _Float16* p = base + (size_t)(col_off + c) * ld + k0 + hs * 16;
  union { v16h v; v8h h[2]; } u;
  u.h[0] = *(const v8h*)p;
  u.h[1] = *(const v8h*)(p + 8);
  return u.v;
}
// B (32x16) from transposed-[n][k] f32 source: 16 contiguous f32 per lane,
// loaded as 4x float4 (ds_load_b128) before converting.
__device__ __forceinline__ v16h ldsBT_frag_f32(const float* base, int ld, int col_off, int k0){
  int lane = threadIdx.x & 31; int hs = lane >> 4; int c = lane & 15;
  const float* p = base + (size_t)(col_off + c) * ld + k0 + hs * 16;
  float4 f0 = *(const float4*)p;
  float4 f1 = *(const float4*)(p + 4);
  float4 f2 = *(const float4*)(p + 8);
  float4 f3 = *(const float4*)(p + 12);
  v16h v;
  v[0]=(_Float16)f0.x;  v[1]=(_Float16)f0.y;  v[2]=(_Float16)f0.z;  v[3]=(_Float16)f0.w;
  v[4]=(_Float16)f1.x;  v[5]=(_Float16)f1.y;  v[6]=(_Float16)f1.z;  v[7]=(_Float16)f1.w;
  v[8]=(_Float16)f2.x;  v[9]=(_Float16)f2.y;  v[10]=(_Float16)f2.z; v[11]=(_Float16)f2.w;
  v[12]=(_Float16)f3.x; v[13]=(_Float16)f3.y; v[14]=(_Float16)f3.z; v[15]=(_Float16)f3.w;
  return v;
}
// B (32x16) from natural-[k][n] f32 source: per-lane ldn-strided (row-parallel
// across lanes -> conflict-free; pairs merge into ds_load_2addr_b32).
__device__ __forceinline__ v16h ldsB_frag_kn_f32(const float* base, int ldn, int col_off, int k0){
  int lane = threadIdx.x & 31; int hs = lane >> 4; int c = lane & 15;
  const float* p = base + (size_t)(k0 + hs * 16) * ldn + col_off + c;
  v16h v;
  #pragma unroll
  for (int j = 0; j < 16; ++j) v[j] = (_Float16)p[(size_t)j * ldn];
  return v;
}

__device__ __forceinline__ v8h pack8(float4 f0, float4 f1){
  v8h h;
  h[0]=(_Float16)f0.x; h[1]=(_Float16)f0.y; h[2]=(_Float16)f0.z; h[3]=(_Float16)f0.w;
  h[4]=(_Float16)f1.x; h[5]=(_Float16)f1.y; h[6]=(_Float16)f1.z; h[7]=(_Float16)f1.w;
  return h;
}

// ---------------------------------------------------------------------------
// TDM 2D tile load (CDNA5 ISA).  Descriptor packing:
// group0: count=1 | lds_addr | global_addr[56:0] | type=2
// group1: data_size=4B, pad fields, tensor_dim0/1, tile_dim0/1, dim0_stride
// Zero-fill beyond tensor dims handles all ragged edges for us.
// ---------------------------------------------------------------------------
__device__ __forceinline__ void tdm_load_2d(unsigned lds_off, const float* gptr,
                                            unsigned dim0, unsigned dim1,
                                            unsigned tile0, unsigned tile1,
                                            unsigned stride0,
                                            unsigned padI, unsigned padA)
{
#if HAVE_TDM
  unsigned long long ga = (unsigned long long)(size_t)(const void*)gptr;
  u32x4 g0;
  g0[0] = 1u;                                           // count=1 (user mode)
  g0[1] = lds_off;                                      // LDS byte address
  g0[2] = (unsigned)(ga & 0xFFFFFFFFull);               // global_addr[31:0]
  g0[3] = (unsigned)((ga >> 32) & 0x01FFFFFFull) | (2u << 30);  // [56:32] | type=2
  unsigned pe = (padI | padA) ? 1u : 0u;
  i32x8 g1;
  g1[0] = (int)((2u << 16) | (pe << 20) | (padI << 22) | (padA << 25)); // 4B elems
  g1[1] = (int)((dim0 & 0xFFFFu) << 16);
  g1[2] = (int)(((dim0 >> 16) & 0xFFFFu) | ((dim1 & 0xFFFFu) << 16));
  g1[3] = (int)(((dim1 >> 16) & 0xFFFFu) | ((tile0 & 0xFFFFu) << 16));
  g1[4] = (int)(tile1 & 0xFFFFu);                       // tile_dim1, tile_dim2=0
  g1[5] = (int)stride0;                                 // dim0_stride[31:0]
  g1[6] = 0; g1[7] = 0;                                 // dim1_stride unused (2D)
  i32x4 z4 = {0, 0, 0, 0};
#if __clang_major__ >= 23
  i32x8 z8 = {0, 0, 0, 0, 0, 0, 0, 0};
  __builtin_amdgcn_tensor_load_to_lds(g0, g1, z4, z4, z8, 0);
#else
  __builtin_amdgcn_tensor_load_to_lds(g0, g1, z4, z4, 0);
#endif
#else
  (void)lds_off; (void)gptr; (void)dim0; (void)dim1; (void)tile0; (void)tile1;
  (void)stride0; (void)padI; (void)padA;
#endif
}
__device__ __forceinline__ void tdm_wait(){
#if HAVE_TDM
  __builtin_amdgcn_s_wait_tensorcnt(0);
#endif
}

// ---------------------------------------------------------------------------
// WMMA GEMM:  out[M,N] = act(A' @ W' + bias) (+ res)
// AMODE 0: A'[m,k]=A[m*lda+k] (TDM, double-buffered)
// AMODE 1: conv gather, K=3*512 (VALU staging; TDM cannot gather taps)
// WMODE 0: W'[k,n]=W[k*N+n] (TDM)    WMODE 1: W'[k,n]=W[(n*512+ci)*3+dt]
// Block 128x128, 8 waves x (64x32), BK=32.  f32 LDS tiles, cvt in frag loads.
// A tile TDM-padded to ld=36 floats -> conflict-free A fragments.
// ---------------------------------------------------------------------------
#define GEMM_SMEM 69632

template<int AMODE, int WMODE>
__global__ void __launch_bounds__(256)
gemm_kernel(const float* __restrict__ A, int lda,
            const float* __restrict__ W,
            const float* __restrict__ bias,
            const float* res, float* out,
            int M, int N, int K, int act, int Tlen)
{
  extern __shared__ char gsm[];
  float* sAF0 = (float*)gsm;                 // 128 x 36
  float* sAF1 = sAF0 + 128 * 36;
  float* sBF0 = (float*)(gsm + 36864);       // 32 x 128  [k][n]
  float* sBF1 = sBF0 + 32 * 128;

  const int tid = threadIdx.x, wid = tid >> 5, lane = tid & 31;
  const int hs = lane >> 4, lc = lane & 15;
  const int m0 = blockIdx.y * 128, n0 = blockIdx.x * 128;
  const int wm = (wid >> 2) * 64, wn = (wid & 3) * 32;

  v8f z = {};
  v8f acc[4][2];
  #pragma unroll
  for (int mi = 0; mi < 4; ++mi){ acc[mi][0] = z; acc[mi][1] = z; }

  auto compute_step = [&](const float* aF, const float* bF){
    v16h af[4];
    #pragma unroll
    for (int mi = 0; mi < 4; ++mi) af[mi] = ldsA_frag_f32(aF, 36, wm + mi * 16, 0);
    #pragma unroll
    for (int ni = 0; ni < 2; ++ni){
      v16h bf = ldsB_frag_kn_f32(bF, 128, wn + ni * 16, 0);
      #pragma unroll
      for (int mi = 0; mi < 4; ++mi) acc[mi][ni] = wmma16(af[mi], bf, acc[mi][ni]);
    }
  };

  const bool useTdm = (AMODE == 0) && (WMODE == 0) && (HAVE_TDM != 0);

  if (useTdm){
    auto issue = [&](int k0, float* aF, float* bF){
      // A tile 128 rows x 32 cols, hw-padded +4 DW per 32-DW row -> ld 36
      tdm_load_2d((unsigned)(size_t)(void*)aF, A + (size_t)m0 * lda + k0,
                  (unsigned)(K - k0), (unsigned)(M - m0), 32u, 128u,
                  (unsigned)lda, 4u, 3u);
      // B tile 32 rows x 128 cols [k][n], no pad (row-parallel frag reads)
      tdm_load_2d((unsigned)(size_t)(void*)bF, W + (size_t)k0 * N + n0,
                  (unsigned)(N - n0), (unsigned)(K - k0), 128u, 32u,
                  (unsigned)N, 0u, 0u);
    };
    const int S = K / 32;
    if (wid == 0) issue(0, sAF0, sBF0);
    for (int s = 0; s < S; ++s){
      float* aF = (s & 1) ? sAF1 : sAF0;
      float* bF = (s & 1) ? sBF1 : sBF0;
      if (wid == 0) tdm_wait();     // current buffers complete
      __syncthreads();              // visible to all; prev compute drained
      if (wid == 0 && s + 1 < S)    // overlap next DMA with this WMMA step
        issue(32 * (s + 1), (s & 1) ? sAF0 : sAF1, (s & 1) ? sBF0 : sBF1);
      compute_step(aF, bF);
    }
  } else {
    for (int k0 = 0; k0 < K; k0 += 32){
      __syncthreads();
      if (k0 + 32 < K)
        __builtin_prefetch(W + (size_t)(k0 + 32) * (WMODE == 0 ? N : 3), 0, 0);
      // ---- stage A tile (128x32) f32
      #pragma unroll
      for (int i = 0; i < 4; ++i){
        int e = tid + i * 256;
        int row = e >> 3, k4 = (e & 7) * 4;
        int gm = m0 + row;
        float4 f = {0.f, 0.f, 0.f, 0.f};
        if (AMODE == 0){
          if (gm < M) f = *(const float4*)(A + (size_t)gm * lda + (k0 + k4));
        } else {
          int gk = k0 + k4;
          int dt = gk >> 9, ci = gk & 511;
          int t  = gm & (Tlen - 1);
          int ts = t + dt - 1;
          if (gm < M && ts >= 0 && ts < Tlen)
            f = *(const float4*)(A + (size_t)(gm + dt - 1) * lda + ci);
        }
        *(float4*)(sAF0 + row * 36 + k4) = f;
      }
      // ---- stage B tile (32x128) f32 [k][n]
      #pragma unroll
      for (int i = 0; i < 4; ++i){
        int e = tid + i * 256;
        int kk = e >> 5, n4 = (e & 31) * 4;
        int gk = k0 + kk;
        float4 o = {0.f, 0.f, 0.f, 0.f};
        if (WMODE == 0){
          int gn = n0 + n4;
          if (gn + 3 < N) o = *(const float4*)(W + (size_t)gk * N + gn);
          else {
            float vv[4] = {0.f, 0.f, 0.f, 0.f};
            #pragma unroll
            for (int j = 0; j < 4; ++j) if (gn + j < N) vv[j] = W[(size_t)gk * N + gn + j];
            o.x = vv[0]; o.y = vv[1]; o.z = vv[2]; o.w = vv[3];
          }
        } else {
          int ci = gk & 511, dt = gk >> 9;
          float vv[4] = {0.f, 0.f, 0.f, 0.f};
          #pragma unroll
          for (int j = 0; j < 4; ++j){
            int gn = n0 + n4 + j;
            if (gn < N) vv[j] = W[((size_t)gn * 512 + ci) * 3 + dt];
          }
          o.x = vv[0]; o.y = vv[1]; o.z = vv[2]; o.w = vv[3];
        }
        *(float4*)(sBF0 + kk * 128 + n4) = o;
      }
      __syncthreads();
      compute_step(sAF0, sBF0);
    }
  }

  // ---- epilogue
  #pragma unroll
  for (int mi = 0; mi < 4; ++mi)
  #pragma unroll
  for (int ni = 0; ni < 2; ++ni){
    int col = n0 + wn + ni * 16 + lc;
    float bv = (col < N) ? bias[col] : 0.f;
    #pragma unroll
    for (int i = 0; i < 8; ++i){
      int row = m0 + wm + mi * 16 + hs * 8 + i;
      if (row < M && col < N){
        float v = acc[mi][ni][i] + bv;
        if (act == ACT_SILU)      v = v / (1.f + __expf(-v));
        else if (act == ACT_RELU) v = fmaxf(v, 0.f);
        else if (act == ACT_SIG)  v = 1.f / (1.f + __expf(-v));
        if (res) v += res[(size_t)row * N + col];
        out[(size_t)row * N + col] = v;
      }
    }
  }
}

// ---------------------------------------------------------------------------
// Attention: block = (b, h, 16-query tile), 4 waves.
// Phase 1: per-wave TDM-staged K chunk (64x64, ld 68 via TDM pad) and rel
//          window (80x64, ld 68) -> WMMA scores + Toeplitz-gathered rel bias.
//          TENSORcnt is per-wave -> zero block barriers in this loop.
// Phase 2: softmax rows in LDS.  Phase 3: O = P @ V via WMMA.
// LDS: 250,112 B (within CDNA5's 320 KB/workgroup).
// ---------------------------------------------------------------------------
#define ATT_SMEM 250112

__global__ void __launch_bounds__(128)
attn_kernel(const float* __restrict__ Q, const float* __restrict__ K,
            const float* __restrict__ V, const float* __restrict__ Rel,
            float* __restrict__ O)
{
  extern __shared__ char smem[];
  const int T = 1024, DM = 512, DH = 64, NREL = 2047;
  const int qt = blockIdx.x, h = blockIdx.y, b = blockIdx.z;
  const int q0 = qt * 16;
  const int tid = threadIdx.x, wid = tid >> 5, lane = tid & 31;
  const int hs = lane >> 4, lc = lane & 15;

  _Float16* sQ  = (_Float16*)(smem);                          // 16x72 f16
  float*    sS  = (float*)(smem + 2304);                      // 16x1024 f32
  float*    sKF = (float*)(smem + 67840)  + wid * (64 * 68);  // per-wave K f32
  float*    sRF = (float*)(smem + 137472) + wid * (80 * 68);  // per-wave rel f32
  float*    sG  = (float*)(smem + 224512) + wid * (16 * 80);  // per-wave G
  _Float16* sVT = (_Float16*)(smem + 244992);                 // shared V^T f16

  { // stage Q tile (16x64) f32 -> f16
    const float* src = Q + ((size_t)(b * T + q0)) * DM + h * DH;
    int r = tid >> 3, c8 = (tid & 7) * 8;
    const float* p = src + r * DM + c8;
    float4 f0 = *(const float4*)p, f1 = *(const float4*)(p + 4);
    *(v8h*)(sQ + r * 72 + c8) = pack8(f0, f1);
  }
  __syncthreads();

  v16h qa0 = ldsA_frag_h(sQ, 72, 0, 0);
  v16h qa1 = ldsA_frag_h(sQ, 72, 0, 32);

  // ---- phase 1
  for (int chunk = wid; chunk < T / 64; chunk += 4){
    int k0 = chunk * 64;
    int w0 = k0 - q0 + (T - 16);          // rel window base (always >= 0)
    const float* kptr = K + ((size_t)(b * T + k0)) * DM + h * DH;
#if HAVE_TDM
    tdm_load_2d((unsigned)(size_t)(void*)sKF, kptr, 64u, 64u, 64u, 64u,
                (unsigned)DM, 5u, 3u);                       // ld 68
    tdm_load_2d((unsigned)(size_t)(void*)sRF, Rel + (size_t)w0 * DH,
                64u, (unsigned)(NREL - w0), 64u, 80u,
                (unsigned)DH, 5u, 3u);                       // ld 68, OOB rows -> 0
    tdm_wait();
    asm volatile("" ::: "memory");
#else
    for (int e = lane; e < 64 * 8; e += 32){
      int r = e >> 3, c8 = (e & 7) * 8;
      const float* p = kptr + r * DM + c8;
      *(float4*)(sKF + r * 68 + c8)     = *(const float4*)p;
      *(float4*)(sKF + r * 68 + c8 + 4) = *(const float4*)(p + 4);
    }
    for (int e = lane; e < 80 * 8; e += 32){
      int r = e >> 3, c8 = (e & 7) * 8;
      int rr = w0 + r;
      float4 f0 = {0.f,0.f,0.f,0.f}, f1 = {0.f,0.f,0.f,0.f};
      if (rr >= 0 && rr < NREL){
        const float* p = Rel + (size_t)rr * DH + c8;
        f0 = *(const float4*)p; f1 = *(const float4*)(p + 4);
      }
      *(float4*)(sRF + r * 68 + c8)     = f0;
      *(float4*)(sRF + r * 68 + c8 + 4) = f1;
    }
    asm volatile("" ::: "memory");
#endif
    v8f zz = {};
    v8f sc[4];
    #pragma unroll
    for (int ni = 0; ni < 4; ++ni){               // main scores 16x64
      v8f a = zz;
      a = wmma16(qa0, ldsBT_frag_f32(sKF, 68, ni * 16, 0),  a);
      a = wmma16(qa1, ldsBT_frag_f32(sKF, 68, ni * 16, 32), a);
      sc[ni] = a;
    }
    #pragma unroll
    for (int ni = 0; ni < 5; ++ni){               // G = Q @ RelWin^T (16x80)
      v8f a = zz;
      a = wmma16(qa0, ldsBT_frag_f32(sRF, 68, ni * 16, 0),  a);
      a = wmma16(qa1, ldsBT_frag_f32(sRF, 68, ni * 16, 32), a);
      #pragma unroll
      for (int i = 0; i < 8; ++i) sG[(hs * 8 + i) * 80 + ni * 16 + lc] = a[i];
    }
    asm volatile("" ::: "memory");                // wave-private; DS in-order
    #pragma unroll
    for (int ni = 0; ni < 4; ++ni)
    #pragma unroll
    for (int i = 0; i < 8; ++i){
      int qi = hs * 8 + i;
      int kc = ni * 16 + lc;
      sS[qi * 1024 + k0 + kc] = sc[ni][i] + sG[qi * 80 + kc + 15 - qi];
    }
  }
  __syncthreads();

  { // ---- phase 2: softmax((S)*1/8) per row; 4 rows per wave
    const float scale = 0.125f;
    for (int r = wid * 4; r < wid * 4 + 4; ++r){
      float* Sr = sS + r * 1024;
      float mx = -3.4e38f;
      for (int j = lane; j < 1024; j += 32) mx = fmaxf(mx, Sr[j]);
      #pragma unroll
      for (int off = 16; off; off >>= 1) mx = fmaxf(mx, __shfl_xor(mx, off, 32));
      float sum = 0.f;
      for (int j = lane; j < 1024; j += 32){
        float e = __expf((Sr[j] - mx) * scale);
        Sr[j] = e; sum += e;
      }
      #pragma unroll
      for (int off = 16; off; off >>= 1) sum += __shfl_xor(sum, off, 32);
      float inv = 1.f / sum;
      for (int j = lane; j < 1024; j += 32) Sr[j] *= inv;
    }
  }
  __syncthreads();

  // ---- phase 3: O = P @ V ; wave wid owns head-dim subtile [wid*16, +16)
  v8f oacc = {};
  for (int kc = 0; kc < T / 32; ++kc){
    const float* src = V + ((size_t)(b * T + kc * 32)) * DM + h * DH;
    for (int e = tid; e < 32 * 16; e += 128){     // V chunk transposed [d][k]
      int kr = e >> 4, d4 = (e & 15) * 4;
      float4 f = *(const float4*)(src + kr * DM + d4);
      sVT[(d4 + 0) * 40 + kr] = (_Float16)f.x;
      sVT[(d4 + 1) * 40 + kr] = (_Float16)f.y;
      sVT[(d4 + 2) * 40 + kr] = (_Float16)f.z;
      sVT[(d4 + 3) * 40 + kr] = (_Float16)f.w;
    }
    __syncthreads();
    v16h pa = ldsA_frag_f32(sS, 1024, 0, kc * 32);
    v16h vb = ldsBT_frag(sVT, 40, wid * 16, 0);
    oacc = wmma16(pa, vb, oacc);
    __syncthreads();
  }
  {
    float* dst = O + ((size_t)(b * T + q0)) * DM + h * DH + wid * 16;
    #pragma unroll
    for (int i = 0; i < 8; ++i) dst[(hs * 8 + i) * DM + lc] = oacc[i];
  }
}

// ---------------------------------------------------------------------------
// Small supporting kernels
// ---------------------------------------------------------------------------
__global__ void __launch_bounds__(256)
ln_kernel(const float* __restrict__ x, const float* __restrict__ gw,
          const float* __restrict__ bw, float* __restrict__ y, int D)
{
  __shared__ float rs[8], rs2[8];
  int row = blockIdx.x;
  const float* xr = x + (size_t)row * D;
  float* yr = y + (size_t)row * D;
  float s = 0.f, s2 = 0.f;
  for (int j = threadIdx.x; j < D; j += blockDim.x){ float v = xr[j]; s += v; s2 += v * v; }
  #pragma unroll
  for (int off = 16; off; off >>= 1){ s += __shfl_down(s, off, 32); s2 += __shfl_down(s2, off, 32); }
  int wid = threadIdx.x >> 5, lane = threadIdx.x & 31;
  if (!lane){ rs[wid] = s; rs2[wid] = s2; }
  __syncthreads();
  if (threadIdx.x == 0){
    float a = 0.f, c = 0.f;
    for (int i = 0; i < (int)(blockDim.x >> 5); ++i){ a += rs[i]; c += rs2[i]; }
    rs[0] = a; rs2[0] = c;
  }
  __syncthreads();
  float mu  = rs[0] / D;
  float var = rs2[0] / D - mu * mu;
  float inv = rsqrtf(var + 1e-5f);
  for (int j = threadIdx.x; j < D; j += blockDim.x)
    yr[j] = (xr[j] - mu) * inv * gw[j] + bw[j];
}

__global__ void rel1_kernel(const float* __restrict__ w, const float* __restrict__ b,
                            const float* __restrict__ a, float* __restrict__ out,
                            int R, int D, int Toff)
{
  int i = blockIdx.x * blockDim.x + threadIdx.x;
  if (i >= R * D) return;
  int r = i / D, j = i - r * D;
  float v = (float)(r - Toff) * w[j] + b[j];
  out[i] = v >= 0.f ? v : a[0] * v;
}

__global__ void rowdot_kernel(const float* __restrict__ g, const float* __restrict__ w,
                              const float* __restrict__ b, float* __restrict__ out, int D)
{
  int m = blockIdx.x, lane = threadIdx.x;
  float s = 0.f;
  for (int j = lane; j < D; j += 32) s += g[(size_t)m * D + j] * w[j];
  #pragma unroll
  for (int off = 16; off; off >>= 1) s += __shfl_down(s, off, 32);
  if (!lane) out[2 * m + 1] = 1.f / (1.f + __expf(-(s + b[0])));
}

__global__ void convfinal_kernel(const float* __restrict__ x, const float* __restrict__ w,
                                 const float* __restrict__ b, float* __restrict__ out, int Tlen)
{
  int m = blockIdx.x, lane = threadIdx.x;
  int t = m & (Tlen - 1);
  float s = 0.f;
  for (int kk = lane; kk < 1536; kk += 32){
    int ci = kk / 3, dt = kk - ci * 3;
    int ts = t + dt - 1;
    if (ts >= 0 && ts < Tlen) s += x[(size_t)(m + dt - 1) * 512 + ci] * w[kk];
  }
  #pragma unroll
  for (int off = 16; off; off >>= 1) s += __shfl_down(s, off, 32);
  if (!lane) out[2 * m] = 1.f / (1.f + __expf(-(s + b[0])));
}

// ---------------------------------------------------------------------------
// Host orchestration
// ---------------------------------------------------------------------------
static inline void launch_gemm_std(hipStream_t s, const float* A, int lda, const float* W,
                                   const float* bias, const float* res, float* out,
                                   int M, int N, int K, int act){
  dim3 g((N + 127) / 128, (M + 127) / 128);
  gemm_kernel<0, 0><<<g, 256, GEMM_SMEM, s>>>(A, lda, W, bias, res, out, M, N, K, act, 1024);
}
static inline void launch_gemm_conv(hipStream_t s, const float* X, const float* Wc,
                                    const float* bias, float* out, int M, int act){
  dim3 g((512 + 127) / 128, (M + 127) / 128);
  gemm_kernel<1, 1><<<g, 256, GEMM_SMEM, s>>>(X, 512, Wc, bias, nullptr, out, M, 512, 1536, act, 1024);
}

extern "C" void kernel_launch(void* const* d_in, const int* in_sizes, int n_in,
                              void* d_out, int out_size, void* d_ws, size_t ws_size,
                              hipStream_t stream)
{
  (void)in_sizes; (void)n_in; (void)out_size; (void)ws_size;
  const float* x = (const float*)d_in[0];
  auto P = [&](int i){ return (const float*)d_in[i]; };
  float* ws  = (float*)d_ws;
  float* out = (float*)d_out;

  const int M = 4096;   // B*T
  const size_t O_H = 0;
  const size_t O_A = O_H + (size_t)M * 512;
  const size_t O_Q = O_A + (size_t)M * 512;
  const size_t O_K = O_Q + (size_t)M * 512;
  const size_t O_V = O_K + (size_t)M * 512;
  const size_t O_O = O_V + (size_t)M * 512;
  const size_t O_T1 = O_O + (size_t)M * 512;     // 4096x2048
  const size_t O_REL = O_T1 + (size_t)M * 2048;  // 2047x64
  float *H = ws + O_H, *Abuf = ws + O_A, *Qb = ws + O_Q, *Kb = ws + O_K;
  float *Vb = ws + O_V, *Ob = ws + O_O, *T1 = ws + O_T1, *REL = ws + O_REL;

  // encoder: h = silu(silu(x@enc1)@enc2)
  launch_gemm_std(stream, x, 256, P(1), P(2), nullptr, Abuf, M, 512, 256, ACT_SILU);
  launch_gemm_std(stream, Abuf, 512, P(3), P(4), nullptr, H, M, 512, 512, ACT_SILU);

  // local conv stack (conv = shifted-gather GEMM in (B*T, C) layout)
  launch_gemm_conv(stream, H,    P(5),  P(6),  Abuf, M, ACT_SILU);
  launch_gemm_conv(stream, Abuf, P(7),  P(8),  Qb,   M, ACT_SILU);
  launch_gemm_conv(stream, Qb,   P(9),  P(10), Abuf, M, ACT_SILU);
  launch_gemm_conv(stream, Abuf, P(11), P(12), Qb,   M, ACT_SILU);
  convfinal_kernel<<<M, 32, 0, stream>>>(Qb, P(13), P(14), out, 1024);  // out[2m]

  // relative-position table: rel = prelu(rp@rel1)@rel2   (2047 x 64)
  rel1_kernel<<<(2047 * 512 + 255) / 256, 256, 0, stream>>>(P(15), P(16), P(17), T1, 2047, 512, 1023);
  launch_gemm_std(stream, T1, 512, P(18), P(19), nullptr, REL, 2047, 64, 512, ACT_NONE);

  // transformer layers (pre-LN), g lives in H
  for (int l = 0; l < 4; ++l){
    int bp = 20 + 16 * l;
    ln_kernel<<<M, 256, 0, stream>>>(H, P(bp + 8), P(bp + 9), Abuf, 512);
    launch_gemm_std(stream, Abuf, 512, P(bp + 0), P(bp + 1), nullptr, Qb, M, 512, 512, ACT_NONE);
    launch_gemm_std(stream, Abuf, 512, P(bp + 2), P(bp + 3), nullptr, Kb, M, 512, 512, ACT_NONE);
    launch_gemm_std(stream, Abuf, 512, P(bp + 4), P(bp + 5), nullptr, Vb, M, 512, 512, ACT_NONE);
    attn_kernel<<<dim3(64, 8, 4), 128, ATT_SMEM, stream>>>(Qb, Kb, Vb, REL, Ob);
    launch_gemm_std(stream, Ob, 512, P(bp + 6), P(bp + 7), H, H, M, 512, 512, ACT_NONE);   // g += o@wo
    ln_kernel<<<M, 256, 0, stream>>>(H, P(bp + 14), P(bp + 15), Abuf, 512);
    launch_gemm_std(stream, Abuf, 512, P(bp + 10), P(bp + 11), nullptr, T1, M, 2048, 512, ACT_RELU);
    launch_gemm_std(stream, T1, 2048, P(bp + 12), P(bp + 13), H, H, M, 512, 2048, ACT_NONE); // g += relu@w2
  }

  rowdot_kernel<<<M, 32, 0, stream>>>(H, P(86), P(87), out, 512);
}
